// MultiHeadAttention_64209761075860
// MI455X (gfx1250) — matrix operand order repaired and checked by reading, not compiled, not run on hardware
//
#include <hip/hip_runtime.h>

typedef __attribute__((ext_vector_type(16))) _Float16 v16h;
typedef __attribute__((ext_vector_type(8)))  _Float16 v8h;
typedef __attribute__((ext_vector_type(8)))  float    v8f;
typedef __attribute__((ext_vector_type(4)))  float    v4f;

static __device__ __forceinline__ v8f wmma_f16(v16h a, v16h b, v8f c) {
  // D = A(16x32 f16) * B(32x16 f16) + C(16x16 f32)
  return __builtin_amdgcn_wmma_f32_16x16x32_f16(false, a, false, b, (short)0, c,
                                                false, false);
}

// A-fragment: lane holds row (lane%16); K elements {kb..kb+7, kb+16..kb+23},
// kb = (lane/16)*8 already folded into `p` by caller.
static __device__ __forceinline__ v16h load_a_frag(const _Float16* p) {
  v8h lo = *(const v8h*)(p);
  v8h hi = *(const v8h*)(p + 16);
  return __builtin_shufflevector(lo, hi, 0, 1, 2, 3, 4, 5, 6, 7,
                                 8, 9, 10, 11, 12, 13, 14, 15);
}

// LDS byte offset of a __shared__ object: flat LDS-aperture addresses carry the
// wave-relative LDS address in bits [31:0] (ISA 10.2).
static __device__ __forceinline__ unsigned lds_off(const void* p) {
  return (unsigned)(unsigned long long)p;
}

// Async DMA: LDS[vdst] = MEM[vaddr], 16 bytes per lane, tracked by ASYNCcnt.
static __device__ __forceinline__ void async_b128(unsigned lds, const void* g) {
  asm volatile("global_load_async_to_lds_b128 %0, %1, off"
               :: "v"(lds), "v"((unsigned long long)g)
               : "memory");
}

// ---------------- fp32 -> fp16 conversion (8 elems/thread) ----------------
__global__ void cvt_f32_to_f16(const float* __restrict__ in,
                               _Float16* __restrict__ out, int n) {
  int i = (blockIdx.x * blockDim.x + threadIdx.x) * 8;
  if (i >= n) return;
  v4f a = *(const v4f*)(in + i);
  v4f b = *(const v4f*)(in + i + 4);
  v8h o;
#pragma unroll
  for (int j = 0; j < 4; ++j) { o[j] = (_Float16)a[j]; o[j + 4] = (_Float16)b[j]; }
  *(v8h*)(out + i) = o;
}

// ---------------- GEMM: Out[M,N] = A[M,K] * B[N,K]^T  (f16 in, f32 acc) ----
// 256 threads = 8 waves as 2(M) x 4(N) over a 64x128 macro-tile.
// Double-buffered 64-wide K slabs in LDS, filled with
// global_load_async_to_lds_b128; 8 WMMAs per synchronization.
template <int OUT_F32>
__global__ __launch_bounds__(256) void gemm_nt_wmma(
    const _Float16* __restrict__ A, const _Float16* __restrict__ B,
    void* __restrict__ outp, int M, int N, int K) {
  __shared__ __align__(16) _Float16 As[2][64 * 64];   // 64 rows x 64 k-halves
  __shared__ __align__(16) _Float16 Bs[2][128 * 64];  // 128 rows x 64 k-halves

  const int tid  = threadIdx.x;
  const int w    = tid >> 5;
  const int lane = tid & 31;
  const int nb   = N >> 7;
  const int bm   = blockIdx.x / nb;
  const int bn   = blockIdx.x - bm * nb;
  const int row0 = bm << 6;
  const int col0 = bn << 7;
  const int wm   = w >> 2;   // 0..1
  const int wn   = w & 3;    // 0..3
  const int r    = lane & 15;
  const int hs   = lane >> 4;

  // copy mapping: 16B chunks over a 64-half slab (8 chunks per row).
  // A: 512 chunks (2/thread), B: 1024 chunks (4/thread).
  const int crow = tid >> 3;           // 0..31
  const int coff = (tid & 7) * 8;      // halves: 0..56
  const _Float16* agp1 = A + (size_t)(row0 + crow) * K + coff;
  const _Float16* agp2 = A + (size_t)(row0 + 32 + crow) * K + coff;
  const _Float16* bgp[4];
  unsigned lA1[2], lA2[2], lB[4][2];
#pragma unroll
  for (int j = 0; j < 4; ++j)
    bgp[j] = B + (size_t)(col0 + 32 * j + crow) * K + coff;
#pragma unroll
  for (int p = 0; p < 2; ++p) {
    lA1[p] = lds_off(&As[p][crow * 64 + coff]);
    lA2[p] = lds_off(&As[p][(32 + crow) * 64 + coff]);
#pragma unroll
    for (int j = 0; j < 4; ++j)
      lB[j][p] = lds_off(&Bs[p][(32 * j + crow) * 64 + coff]);
  }

  v8f c00 = {}, c01 = {}, c10 = {}, c11 = {};

  // prologue: fill buffer 0 with k-slab 0
  async_b128(lA1[0], agp1);
  async_b128(lA2[0], agp2);
#pragma unroll
  for (int j = 0; j < 4; ++j) async_b128(lB[j][0], bgp[j]);

  const int nsteps = K >> 6;  // 64-wide slabs
  int buf = 0;
#pragma unroll 2
  for (int s = 0; s < nsteps; ++s) {
    const bool haveNext = (s + 1 < nsteps);
    if (haveNext) {
      const int k1 = (s + 1) << 6;
      async_b128(lA1[buf ^ 1], agp1 + k1);
      async_b128(lA2[buf ^ 1], agp2 + k1);
#pragma unroll
      for (int j = 0; j < 4; ++j) async_b128(lB[j][buf ^ 1], bgp[j] + k1);
      // 12 in flight; in-order completion => <=6 means this slab's 6 landed
      asm volatile("s_wait_asynccnt 6" ::: "memory");
    } else {
      asm volatile("s_wait_asynccnt 0" ::: "memory");
    }
    __syncthreads();

#pragma unroll
    for (int ks = 0; ks < 64; ks += 32) {
      const _Float16* as = &As[buf][(wm * 32 + r) * 64 + ks + hs * 8];
      const _Float16* bs = &Bs[buf][(wn * 32 + r) * 64 + ks + hs * 8];
      v16h af0 = load_a_frag(as);
      v16h af1 = load_a_frag(as + 16 * 64);
      v16h bf0 = load_a_frag(bs);
      v16h bf1 = load_a_frag(bs + 16 * 64);
      c00 = wmma_f16(af0, bf0, c00);
      c01 = wmma_f16(af0, bf1, c01);
      c10 = wmma_f16(af1, bf0, c10);
      c11 = wmma_f16(af1, bf1, c11);
    }

    __syncthreads();  // all waves done reading before buffer is overwritten
    buf ^= 1;
  }

  // C/D layout: elem v of lane = D[v + 8*(lane/16)][lane%16]
  const int col  = col0 + (wn << 5) + r;
  const int rowb = row0 + (wm << 5) + hs * 8;
  if (OUT_F32) {
    float* O = (float*)outp;
#pragma unroll
    for (int v = 0; v < 8; ++v) {
      O[(size_t)(rowb + v) * N + col]           = c00[v];
      O[(size_t)(rowb + v) * N + col + 16]      = c01[v];
      O[(size_t)(rowb + v + 16) * N + col]      = c10[v];
      O[(size_t)(rowb + v + 16) * N + col + 16] = c11[v];
    }
  } else {
    _Float16* O = (_Float16*)outp;
#pragma unroll
    for (int v = 0; v < 8; ++v) {
      O[(size_t)(rowb + v) * N + col]           = (_Float16)c00[v];
      O[(size_t)(rowb + v) * N + col + 16]      = (_Float16)c01[v];
      O[(size_t)(rowb + v + 16) * N + col]      = (_Float16)c10[v];
      O[(size_t)(rowb + v + 16) * N + col + 16] = (_Float16)c11[v];
    }
  }
}

// ---------------- per-token RoPE + head-attention ----------------
static __device__ __forceinline__ void rope_pair(v16h& f, int e, int i, float pos) {
  // inv_freq = 10000^(-i/32), matches reference arange(0,64,2)/64
  float ang = pos * (1.0f / powf(10000.0f, (float)i * (1.0f / 32.0f)));
  float sn = sinf(ang), cs = cosf(ang);
  float x1 = (float)f[e], x2 = (float)f[e + 1];
  f[e]     = (_Float16)(x1 * cs - x2 * sn);
  f[e + 1] = (_Float16)(x2 * cs + x1 * sn);
}

// One wave per token. qkv row (3072 f16) = [q(16x64) | k(16x64) | v(16x64)].
// scores(16x16) via WMMA, softmax via ds_bpermute shuffles within 16-lane
// halves, attn@v via LDS, output scattered per transpose(0,2,1,3).reshape.
__global__ __launch_bounds__(256) void attn_rope_softmax(
    const _Float16* __restrict__ qkv, _Float16* __restrict__ y) {
  __shared__ __align__(16) float    attn_s[8][256];
  __shared__ __align__(16) _Float16 v_s[8][1024];

  const int w    = threadIdx.x >> 5;
  const int lane = threadIdx.x & 31;
  const int tok  = blockIdx.x * 8 + w;   // grid is exact: 16384 tokens
  const int b    = tok >> 13;
  const int tt   = tok & 8191;
  const float pos = (float)tt;
  const int r  = lane & 15;
  const int hs = lane >> 4;

  const _Float16* base = qkv + (size_t)tok * 3072;

  // async-stage v (16x64 f16) into LDS while we do RoPE + scores
  {
    const _Float16* vsrc = base + 2048 + lane * 32;
#pragma unroll
    for (int c = 0; c < 4; ++c)
      async_b128(lds_off(&v_s[w][lane * 32 + c * 8]), vsrc + c * 8);
  }

  const _Float16* qrow = base + r * 64 + hs * 8;          // A layout base
  const _Float16* krow = base + 1024 + r * 64 + hs * 8;   // B layout base

  v16h qf[2], kf[2];
#pragma unroll
  for (int j = 0; j < 2; ++j) {
    qf[j] = load_a_frag(qrow + 32 * j);
    kf[j] = load_a_frag(krow + 32 * j);
    const int dbA = 32 * j + hs * 8;   // A frag: d = dbA + {0..7,16..23}
    const int dbB = 32 * j + hs * 8;   // B frag: d = dbB + {0..7,16..23}
#pragma unroll
    for (int p = 0; p < 8; ++p) {
      rope_pair(qf[j], 2 * p, (dbA >> 1) + p + (p >= 4 ? 4 : 0), pos);
      rope_pair(kf[j], 2 * p, (dbB >> 1) + p + (p >= 4 ? 4 : 0), pos);
    }
  }

  v8f sc = {};
  sc = wmma_f16(qf[0], kf[0], sc);
  sc = wmma_f16(qf[1], kf[1], sc);

  // softmax over key-heads: row m = v + 8*hs lives across one 16-lane half
#pragma unroll
  for (int v = 0; v < 8; ++v) {
    float val = sc[v] * 0.125f;  // 1/sqrt(64)
    float mx = val;
#pragma unroll
    for (int m = 1; m < 16; m <<= 1) mx = fmaxf(mx, __shfl_xor(mx, m, 32));
    float e = __expf(val - mx);
    float sm = e;
#pragma unroll
    for (int m = 1; m < 16; m <<= 1) sm += __shfl_xor(sm, m, 32);
    attn_s[w][(v + 8 * hs) * 16 + r] = e / sm;
  }

  asm volatile("s_wait_asynccnt 0" ::: "memory");
  __syncthreads();

  // out[m][d] = sum_h attn[m][h] * v[h][d]; lane: m = r, d in [hs*32, hs*32+32)
  const int m_q   = r;
  const int dbase = hs * 32;
  float o[32];
#pragma unroll
  for (int j = 0; j < 32; ++j) o[j] = 0.0f;
  for (int h = 0; h < 16; ++h) {
    float a = attn_s[w][m_q * 16 + h];
#pragma unroll
    for (int c = 0; c < 4; ++c) {
      v8h vv = *(const v8h*)&v_s[w][h * 64 + dbase + c * 8];
#pragma unroll
      for (int j = 0; j < 8; ++j) o[c * 8 + j] += a * (float)vv[j];
    }
  }

  // reference's transpose(0,2,1,3).reshape(B,T,C) scatter:
  // row = b*8192 + m*512 + (tt>>4), col = (tt&15)*64 + d
  const int outrow = (b << 13) + (m_q << 9) + (tt >> 4);
  _Float16* yp = y + (size_t)outrow * 1024 + ((tt & 15) << 6) + dbase;
#pragma unroll
  for (int c = 0; c < 4; ++c) {
    v8h st;
#pragma unroll
    for (int j = 0; j < 8; ++j) st[j] = (_Float16)o[c * 8 + j];
    *(v8h*)(yp + c * 8) = st;
  }
}

extern "C" void kernel_launch(void* const* d_in, const int* in_sizes, int n_in,
                              void* d_out, int out_size, void* d_ws, size_t ws_size,
                              hipStream_t stream) {
  (void)in_sizes; (void)n_in; (void)out_size; (void)ws_size;
  const float* x    = (const float*)d_in[0];
  const float* Wqkv = (const float*)d_in[1];
  const float* Wout = (const float*)d_in[2];
  float* out = (float*)d_out;

  const int BT = 16384, C = 1024, N1 = 3072;
  _Float16* xh    = (_Float16*)d_ws;                 // 32 MB
  _Float16* wqkvh = xh + (size_t)BT * C;             //  6 MB
  _Float16* wouth = wqkvh + (size_t)N1 * C;          //  2 MB
  _Float16* qkvh  = wouth + (size_t)C * C;           // 96 MB
  _Float16* yh    = qkvh + (size_t)BT * N1;          // 32 MB  (~168 MB total)

  cvt_f32_to_f16<<<(BT * C) / (8 * 256), 256, 0, stream>>>(x, xh, BT * C);
  cvt_f32_to_f16<<<(N1 * C) / (8 * 256), 256, 0, stream>>>(Wqkv, wqkvh, N1 * C);
  cvt_f32_to_f16<<<(C * C) / (8 * 256), 256, 0, stream>>>(Wout, wouth, C * C);

  // QKV projection: (16384x1024) @ (3072x1024)^T -> f16
  gemm_nt_wmma<0><<<(BT / 64) * (N1 / 128), 256, 0, stream>>>(
      xh, wqkvh, (void*)qkvh, BT, N1, C);

  // per-token RoPE + head attention
  attn_rope_softmax<<<BT / 8, 256, 0, stream>>>(qkvh, yh);

  // output projection: (16384x1024) @ (1024x1024)^T -> f32
  gemm_nt_wmma<1><<<(BT / 64) * (C / 128), 256, 0, stream>>>(
      yh, wouth, (void*)out, BT, C, C);
}
